// SpatialAttention_60318520705148
// MI455X (gfx1250) — compile-verified, matrix operand order checked by reading, not run on hardware
//
#include <hip/hip_runtime.h>

// ---------------------------------------------------------------------------
// SpatialAttention on MI455X (gfx1250): f16-in / f32-accumulate WMMA pipeline.
//   ~39 GFLOP total, ~75MB min HBM traffic -> bandwidth bound; store all
//   intermediates in f16, keep K/V of each head resident in LDS (320KB/WGP),
//   and do every matmul with v_wmma_f32_16x16x32_f16 (d=32 == WMMA K).
//   Softmax is two-pass (K is LDS-resident so S recompute costs only WMMAs),
//   which removes per-iteration cross-lane ds_bpermute reductions.
// ---------------------------------------------------------------------------

typedef __attribute__((ext_vector_type(16))) _Float16 v16h;
typedef __attribute__((ext_vector_type(8)))  _Float16 v8h;
typedef __attribute__((ext_vector_type(8)))  float    v8f;

#define BB   4
#define TT   12
#define NNODE 512
#define HK   8
#define HD   32
#define DD   256
#define PTOT (BB*TT*NNODE)   /* 24576 rows */
#define KIN  (2*DD)          /* 512  */
#define QKVW (3*DD)          /* 768  */

union H16u { v16h v; v8h h[2]; };

// Load a v_wmma f16 A/B fragment: per-lane two contiguous 8-halfword chunks at
// rowbase+h and rowbase+16+h (h = 8*(lane>=16)). 16B-aligned -> ds/global b128.
__device__ __forceinline__ v16h frag16(const _Float16* rowbase, int h) {
  H16u u;
  u.h[0] = *(const v8h*)(rowbase + h);
  u.h[1] = *(const v8h*)(rowbase + 16 + h);
  return u.v;
}

// ------------------------------- converters --------------------------------

__global__ void cvt_x_kernel(const float* __restrict__ X,
                             const float* __restrict__ STE,
                             _Float16* __restrict__ Xc) {
  int idx = blockIdx.x * 256 + threadIdx.x;       // < PTOT*KIN
  int p = idx >> 9, c = idx & 511;
  float v = (c < DD) ? X[p * DD + c] : STE[p * DD + (c - DD)];
  Xc[idx] = (_Float16)v;
}

__global__ void cvt_wqkv_kernel(const float* __restrict__ Wq,
                                const float* __restrict__ Wk,
                                const float* __restrict__ Wv,
                                const float* __restrict__ bq,
                                const float* __restrict__ bk,
                                const float* __restrict__ bv,
                                _Float16* __restrict__ Wc,
                                float* __restrict__ bc) {
  int idx = blockIdx.x * 256 + threadIdx.x;       // < KIN*QKVW
  int f = idx / QKVW, c = idx - f * QKVW;
  float v;
  if (c < DD)            v = Wq[f * DD + c];
  else if (c < 2 * DD)   v = Wk[f * DD + (c - DD)];
  else                   v = Wv[f * DD + (c - 2 * DD)];
  Wc[idx] = (_Float16)v;
  if (f == 0) {
    float b = (c < DD) ? bq[c] : (c < 2 * DD) ? bk[c - DD] : bv[c - 2 * DD];
    bc[c] = b;
  }
}

__global__ void cvt_wf_kernel(const float* __restrict__ Wf1,
                              const float* __restrict__ Wf2,
                              _Float16* __restrict__ W1h,
                              _Float16* __restrict__ W2h) {
  int idx = blockIdx.x * 256 + threadIdx.x;       // < 2*DD*DD
  if (idx < DD * DD) W1h[idx] = (_Float16)Wf1[idx];
  else               W2h[idx - DD * DD] = (_Float16)Wf2[idx - DD * DD];
}

// ------------------------- generic WMMA GEMM -------------------------------
// C[M,Ncols] = act(A[M,Kdim] @ W[Kdim,Ncols] + bias).  BM=128 BN=64 BK=32.
// 256 threads = 8 waves; wave w owns 16 rows x 64 cols (4 accum tiles).
// W tile staged transposed so its B-fragment loads are contiguous b128s.

#define LDA 40   /* padded f16 stride: 80B rows, 16B aligned, bank-spread */

template <bool RELU, bool F32OUT>
__global__ __launch_bounds__(256)
void gemm_f16_kernel(const _Float16* __restrict__ A,
                     const _Float16* __restrict__ W,
                     const float* __restrict__ bias,
                     _Float16* __restrict__ outH,
                     float* __restrict__ outF,
                     int M, int Kdim, int Ncols) {
  __shared__ _Float16 Al[128 * LDA];
  __shared__ _Float16 Wt[64 * LDA];
  const int nbn = Ncols >> 6;
  const int bm = blockIdx.x / nbn, bn = blockIdx.x - bm * nbn;
  const int m0 = bm << 7, n0 = bn << 6;
  const int tid = threadIdx.x, lane = tid & 31, wv = tid >> 5;
  const int hh = (lane >> 4) << 3;   // 0 | 8
  const int lm = lane & 15;

  v8f acc[4] = {};
  const int nk = Kdim >> 5;
  for (int ks = 0; ks < nk; ++ks) {
    const int k0 = ks << 5;
    // stage A tile 128x32 (16B vector chunks)
    for (int i = tid; i < 512; i += 256) {
      int r = i >> 2, c = i & 3;
      if (ks + 1 < nk)
        __builtin_prefetch(A + (size_t)(m0 + r) * Kdim + k0 + 32 + c * 8, 0, 1);
      *(v8h*)&Al[r * LDA + c * 8] =
          *(const v8h*)(A + (size_t)(m0 + r) * Kdim + k0 + c * 8);
    }
    // stage W tile 32x64 transposed -> Wt[64][LDA]
    for (int i = tid; i < 2048; i += 256) {
      int k = i >> 6, n = i & 63;
      Wt[n * LDA + k] = W[(size_t)(k0 + k) * Ncols + n0 + n];
    }
    __syncthreads();
    v16h af = frag16(&Al[(wv * 16 + lm) * LDA], hh);
#pragma unroll
    for (int nt = 0; nt < 4; ++nt) {
      v16h bf = frag16(&Wt[(nt * 16 + lm) * LDA], hh);
      acc[nt] = __builtin_amdgcn_wmma_f32_16x16x32_f16(
          false, af, false, bf, (short)0, acc[nt], false, false);
    }
    __syncthreads();
  }
  // epilogue: C layout -> row = r + 8*(lane>=16), col = tile*16 + lane%16
  const int rbase = m0 + wv * 16 + hh;
#pragma unroll
  for (int nt = 0; nt < 4; ++nt) {
    int col = n0 + nt * 16 + lm;
    float b = bias[col];
#pragma unroll
    for (int r = 0; r < 8; ++r) {
      float v = acc[nt][r] + b;
      if (RELU) v = fmaxf(v, 0.f);
      size_t off = (size_t)(rbase + r) * Ncols + col;
      if (F32OUT) outF[off] = v;
      else        outH[off] = (_Float16)v;
    }
  }
}

// --------------------------- attention (2-pass) ----------------------------
// One workgroup per (b,t,head,qblock of 128). Whole K_h (512x32, row major)
// and V_h (transposed, 32x512) resident in LDS. Each wave: 16 query rows.
// Pass 1: per-lane row maxima via WMMA S-tiles (no cross-lane work in loop),
// single shuffle reduction after. Pass 2: recompute S, exact softmax (exp
// args <= 0), per-lane partial row sums, P->LDS->A-frag, PV WMMA accumulate.

#define KSTR 40            /* K_h row stride (f16) */
#define VSTR 520           /* V^T row stride (f16): 1040B = 65*16B, aligned */
#define ATTN_LDS_BYTES ((NNODE*KSTR + HD*VSTR + 8*16*KSTR) * 2)  /* 84480 */

__device__ __forceinline__ float red16_max(float v) {
  v = fmaxf(v, __shfl_xor(v, 1, 32));
  v = fmaxf(v, __shfl_xor(v, 2, 32));
  v = fmaxf(v, __shfl_xor(v, 4, 32));
  v = fmaxf(v, __shfl_xor(v, 8, 32));
  return v;
}
__device__ __forceinline__ float red16_sum(float v) {
  v += __shfl_xor(v, 1, 32);
  v += __shfl_xor(v, 2, 32);
  v += __shfl_xor(v, 4, 32);
  v += __shfl_xor(v, 8, 32);
  return v;
}

__global__ __launch_bounds__(256)
void attn_kernel(const _Float16* __restrict__ QKV, _Float16* __restrict__ AO) {
  extern __shared__ _Float16 sm[];
  _Float16* Kl = sm;                       // [512][KSTR]
  _Float16* Vt = sm + NNODE * KSTR;        // [32][VSTR]
  _Float16* Pl = Vt + HD * VSTR;           // [8 waves][16][KSTR]

  const int blk = blockIdx.x;
  const int qb = blk & 3, head = (blk >> 2) & 7, bt = blk >> 5;
  const int pbase = bt * NNODE;
  const int kcol = DD + head * HD, vcol = 2 * DD + head * HD;
  const int tid = threadIdx.x, lane = tid & 31, wv = tid >> 5;
  const int hh = (lane >> 4) << 3, lm = lane & 15;

  // stage K_h: 512 rows x 4 chunks of 8 halfwords
  for (int i = tid; i < 2048; i += 256) {
    int j = i >> 2, c = i & 3;
    *(v8h*)&Kl[j * KSTR + c * 8] =
        *(const v8h*)(QKV + (size_t)(pbase + j) * QKVW + kcol + c * 8);
  }
  // stage V_h transposed
  for (int i = tid; i < NNODE * HD; i += 256) {
    int j = i >> 5, dcol = i & 31;
    Vt[dcol * VSTR + j] = QKV[(size_t)(pbase + j) * QKVW + vcol + dcol];
  }
  __syncthreads();

  const int q0 = qb * 128 + wv * 16;
  v16h aq = frag16(QKV + (size_t)(pbase + q0 + lm) * QKVW + head * HD, hh);
  const float SC = 0.17677669529663687f;   // 1/sqrt(32)

  // ---- pass 1: raw-score row maxima (per-lane, then one reduction) ----
  float mraw[8];
#pragma unroll
  for (int r = 0; r < 8; ++r) mraw[r] = -3.0e38f;
  for (int jt = 0; jt < NNODE / 32; ++jt) {
    const int j0 = jt << 5;
    v16h bk0 = frag16(&Kl[(j0 + lm) * KSTR], hh);
    v16h bk1 = frag16(&Kl[(j0 + 16 + lm) * KSTR], hh);
    v8f z = {};
    v8f s0 = __builtin_amdgcn_wmma_f32_16x16x32_f16(false, aq, false, bk0,
                                                    (short)0, z, false, false);
    v8f s1 = __builtin_amdgcn_wmma_f32_16x16x32_f16(false, aq, false, bk1,
                                                    (short)0, z, false, false);
#pragma unroll
    for (int r = 0; r < 8; ++r)
      mraw[r] = fmaxf(mraw[r], fmaxf(s0[r], s1[r]));
  }
#pragma unroll
  for (int r = 0; r < 8; ++r) mraw[r] = red16_max(mraw[r]);  // row-uniform

  // ---- pass 2: exact softmax + PV accumulation ----
  v8f o0 = {}, o1 = {};
  float lacc[8];
#pragma unroll
  for (int r = 0; r < 8; ++r) lacc[r] = 0.f;

  for (int jt = 0; jt < NNODE / 32; ++jt) {
    const int j0 = jt << 5;
    v16h bk0 = frag16(&Kl[(j0 + lm) * KSTR], hh);
    v16h bk1 = frag16(&Kl[(j0 + 16 + lm) * KSTR], hh);
    v8f z = {};
    v8f s0 = __builtin_amdgcn_wmma_f32_16x16x32_f16(false, aq, false, bk0,
                                                    (short)0, z, false, false);
    v8f s1 = __builtin_amdgcn_wmma_f32_16x16x32_f16(false, aq, false, bk1,
                                                    (short)0, z, false, false);
#pragma unroll
    for (int r = 0; r < 8; ++r) {
      float p0 = __expf((s0[r] - mraw[r]) * SC);   // <= 1, no overflow
      float p1 = __expf((s1[r] - mraw[r]) * SC);
      lacc[r] += p0 + p1;
      int row = r + hh;                            // C-layout row for this lane
      Pl[(wv * 16 + row) * KSTR + lm]      = (_Float16)p0;
      Pl[(wv * 16 + row) * KSTR + 16 + lm] = (_Float16)p1;
    }
    asm volatile("s_wait_dscnt 0" ::: "memory");   // P store -> A-frag load
    v16h ap  = frag16(&Pl[(wv * 16 + lm) * KSTR], hh);
    v16h bv0 = frag16(&Vt[lm * VSTR + j0], hh);
    v16h bv1 = frag16(&Vt[(16 + lm) * VSTR + j0], hh);
    o0 = __builtin_amdgcn_wmma_f32_16x16x32_f16(false, ap, false, bv0,
                                                (short)0, o0, false, false);
    o1 = __builtin_amdgcn_wmma_f32_16x16x32_f16(false, ap, false, bv1,
                                                (short)0, o1, false, false);
  }

#pragma unroll
  for (int r = 0; r < 8; ++r) {
    float inv = 1.f / red16_sum(lacc[r]);
    int row = r + hh;
    size_t p = (size_t)(pbase + q0 + row);
    AO[p * DD + head * HD + lm]      = (_Float16)(o0[r] * inv);
    AO[p * DD + head * HD + 16 + lm] = (_Float16)(o1[r] * inv);
  }
}

// ------------------------------- launcher ----------------------------------

extern "C" void kernel_launch(void* const* d_in, const int* in_sizes, int n_in,
                              void* d_out, int out_size, void* d_ws,
                              size_t ws_size, hipStream_t stream) {
  (void)in_sizes; (void)n_in; (void)out_size; (void)ws_size;
  const float* X   = (const float*)d_in[0];
  const float* STE = (const float*)d_in[1];
  const float* Wq  = (const float*)d_in[2];
  const float* bq  = (const float*)d_in[3];
  const float* Wk  = (const float*)d_in[4];
  const float* bk  = (const float*)d_in[5];
  const float* Wv  = (const float*)d_in[6];
  const float* bv  = (const float*)d_in[7];
  const float* Wf1 = (const float*)d_in[8];
  const float* bf1 = (const float*)d_in[9];
  const float* Wf2 = (const float*)d_in[10];
  const float* bf2 = (const float*)d_in[11];
  float* out = (float*)d_out;

  char* ws = (char*)d_ws;
  size_t off = 0;
  auto take = [&](size_t bytes) {
    size_t o = off;
    off = (off + bytes + 255) & ~(size_t)255;
    return o;
  };
  _Float16* Xc   = (_Float16*)(ws + take((size_t)PTOT * KIN * 2));
  _Float16* Wc   = (_Float16*)(ws + take((size_t)KIN * QKVW * 2));
  float*    bc   = (float*)   (ws + take((size_t)QKVW * 4));
  _Float16* W1h  = (_Float16*)(ws + take((size_t)DD * DD * 2));
  _Float16* W2h  = (_Float16*)(ws + take((size_t)DD * DD * 2));
  _Float16* QKV  = (_Float16*)(ws + take((size_t)PTOT * QKVW * 2));
  _Float16* AOb  = (_Float16*)(ws + take((size_t)PTOT * DD * 2));
  _Float16* F1b  = (_Float16*)(ws + take((size_t)PTOT * DD * 2));

  hipFuncSetAttribute((const void*)attn_kernel,
                      hipFuncAttributeMaxDynamicSharedMemorySize,
                      ATTN_LDS_BYTES);

  // 1) pack/convert to f16
  cvt_x_kernel<<<(PTOT * KIN) / 256, 256, 0, stream>>>(X, STE, Xc);
  cvt_wqkv_kernel<<<(KIN * QKVW) / 256, 256, 0, stream>>>(Wq, Wk, Wv, bq, bk,
                                                          bv, Wc, bc);
  cvt_wf_kernel<<<(2 * DD * DD) / 256, 256, 0, stream>>>(Wf1, Wf2, W1h, W2h);

  // 2) fused QKV projection: [24576,512] x [512,768] + bias, relu
  gemm_f16_kernel<true, false><<<(PTOT / 128) * (QKVW / 64), 256, 0, stream>>>(
      Xc, Wc, bc, QKV, nullptr, PTOT, KIN, QKVW);

  // 3) attention: 384 (b,t,head) x 4 q-blocks
  attn_kernel<<<BB * TT * HK * 4, 256, ATTN_LDS_BYTES, stream>>>(QKV, AOb);

  // 4) FC1 (relu) then FC2 (f32 out)
  gemm_f16_kernel<true, false><<<(PTOT / 128) * (DD / 64), 256, 0, stream>>>(
      AOb, W1h, bf1, F1b, nullptr, PTOT, DD, DD);
  gemm_f16_kernel<false, true><<<(PTOT / 128) * (DD / 64), 256, 0, stream>>>(
      F1b, W2h, bf2, nullptr, out, PTOT, DD, DD);
}
